// Transformer_24790551233211
// MI455X (gfx1250) — matrix operand order repaired and checked by reading, not compiled, not run on hardware
//
#include <hip/hip_runtime.h>
#include <math.h>

// ---------------------------------------------------------------------------
// Transformer fwd for MI455X (gfx1250, wave32, WMMA + async LDS copies).
// GEMM operands are pre-converted to f16 (flat or transposed so both A and B
// are row-major along K), then tiles are staged into LDS with
// GLOBAL_LOAD_ASYNC_TO_LDS_B128 (ASYNCcnt) and consumed by
// V_WMMA_F32_16X16X32_F16 with f32 accumulation.
// ---------------------------------------------------------------------------

typedef __attribute__((ext_vector_type(16))) _Float16 v16h;
typedef __attribute__((ext_vector_type(8)))  float    v8f;

union FragH {            // one WMMA 16-bit operand: 8 dwords = 16 halves
  v16h     h;
  unsigned u[8];
};
union PackH {
  _Float16 h[2];
  unsigned u;
};

static __device__ inline unsigned pk2(float a, float b) {
  PackH t;
  t.h[0] = (_Float16)a;   // low 16 bits  = K
  t.h[1] = (_Float16)b;   // high 16 bits = K+1
  return t.u;
}

static __device__ inline v8f wmma_f16(const v16h& a, const v16h& b, const v8f& c) {
  return __builtin_amdgcn_wmma_f32_16x16x32_f16(false, a, false, b, (short)0, c,
                                                false, false);
}

// async copy 16B global -> LDS (tracked by ASYNCcnt)
static __device__ inline void async_b128(unsigned lds_off, const void* gaddr) {
  asm volatile("global_load_async_to_lds_b128 %0, %1, off"
               :: "v"(lds_off), "v"(gaddr) : "memory");
}
static __device__ inline void wait_async0() {
  asm volatile("s_wait_asynccnt 0x0" ::: "memory");
}

#define Dm   512
#define Hh   8
#define DHd  64
#define Bb   8
#define Tt   512           // TS == TT == 512
#define NLl  6
#define Vv   32000
#define NEG_BIG (-1e30f)

// --------------------------- embedding + posenc ----------------------------
__global__ void embed_kernel(const int* __restrict__ ids,
                             const float* __restrict__ emb,
                             float* __restrict__ out) {
  const int row = blockIdx.x;            // b*T + t
  const int t   = row & (Tt - 1);
  const int id  = ids[row];
  const float scale = 22.62741699796952f;      // sqrt(512)
  for (int d = threadIdx.x; d < Dm; d += blockDim.x) {
    float e = emb[(size_t)id * Dm + d] * scale;
    float freq = __expf(-9.210340371976184f * (float)(d & ~1) * (1.0f / Dm));
    float ang  = (float)t * freq;
    out[(size_t)row * Dm + d] = e + ((d & 1) ? __cosf(ang) : __sinf(ang));
  }
}

// ------------------------ fused add + LayerNorm ----------------------------
__global__ void ln_add_kernel(const float* __restrict__ x,
                              const float* __restrict__ r,
                              const float* __restrict__ g,
                              const float* __restrict__ bta,
                              float* __restrict__ out) {
  __shared__ float red[256];
  const int row = blockIdx.x, tid = threadIdx.x;
  const float* xr = x + (size_t)row * Dm;
  const float* rr = r + (size_t)row * Dm;
  float v0 = xr[tid] + rr[tid];
  float v1 = xr[tid + 256] + rr[tid + 256];
  red[tid] = v0 + v1;
  __syncthreads();
  for (int o = 128; o > 0; o >>= 1) {
    if (tid < o) red[tid] += red[tid + o];
    __syncthreads();
  }
  const float mean = red[0] * (1.0f / Dm);
  __syncthreads();
  const float d0 = v0 - mean, d1 = v1 - mean;
  red[tid] = d0 * d0 + d1 * d1;
  __syncthreads();
  for (int o = 128; o > 0; o >>= 1) {
    if (tid < o) red[tid] += red[tid + o];
    __syncthreads();
  }
  const float rstd = rsqrtf(red[0] * (1.0f / Dm) + 1e-5f);
  out[(size_t)row * Dm + tid]       = d0 * rstd * g[tid] + bta[tid];
  out[(size_t)row * Dm + tid + 256] = d1 * rstd * g[tid + 256] + bta[tid + 256];
}

// ------------------------- operand conversion ------------------------------
// flat fp32 -> f16 (layout preserved); n multiple of 1024
__global__ void cvt_kernel(const float* __restrict__ src,
                           _Float16* __restrict__ dst) {
  const size_t idx = ((size_t)blockIdx.x * 256 + threadIdx.x) * 4;
  float4 v = *reinterpret_cast<const float4*>(src + idx);
  uint2 p;
  p.x = pk2(v.x, v.y);
  p.y = pk2(v.z, v.w);
  *reinterpret_cast<uint2*>(dst + idx) = p;
}

// fp32 [K,N] -> f16 [N,K] (LDS-tiled transpose); K,N multiples of 32
__global__ void cvt_tr_kernel(const float* __restrict__ src,
                              _Float16* __restrict__ dst, int K, int N) {
  __shared__ float tile[32][33];
  const int kb = blockIdx.y * 32, nb = blockIdx.x * 32;
  const int tx = threadIdx.x & 31, ty = threadIdx.x >> 5;   // 32x8
#pragma unroll
  for (int i = 0; i < 4; ++i)
    tile[ty + i * 8][tx] = src[(size_t)(kb + ty + i * 8) * N + nb + tx];
  __syncthreads();
#pragma unroll
  for (int i = 0; i < 4; ++i)
    dst[(size_t)(nb + ty + i * 8) * K + kb + tx] =
        (_Float16)tile[tx][ty + i * 8];
}

// ------------------------------ WMMA GEMM ----------------------------------
// C[M,N] = act(A16[M,K] @ Bt16^T + bias), A16 row-major [M,K] f16,
// Bt16 row-major [N,K] f16.  Block: 256 threads (8 wave32), tile 128x128x32;
// wave (wm,wn) owns 32x64.  Tiles staged via async global->LDS B128 copies.
// LDS row stride 20 dwords: 16B-aligned rows, conflict-free column reads.
#define BM 128
#define BN 128
#define BK 32
#define KDW 20

__global__ __launch_bounds__(256)
void gemm_kernel(const _Float16* __restrict__ A16,
                 const _Float16* __restrict__ Bt16,
                 const float* __restrict__ bias, int relu,
                 float* __restrict__ C, int M, int N, int K) {
  __shared__ unsigned As[BM][KDW];
  __shared__ unsigned Bs[BN][KDW];

  const int m0 = blockIdx.y * BM;
  const int n0 = blockIdx.x * BN;
  const int tid  = threadIdx.x;
  const int lane = tid & 31;
  const int wv   = tid >> 5;
  const int wm   = wv & 3;            // 4 waves along M
  const int wn   = wv >> 2;           // 2 waves along N
  const int half = lane >> 4;
  const int l16  = lane & 15;

  v8f acc[2][4] = {};

  const int nk = K / BK;
  for (int kt = 0; kt < nk; ++kt) {
    const int k0 = kt * BK;
    __syncthreads();                 // all waves done reading previous tile
    // ---- async-stage A and B tiles (each: 128 rows x 32 halves = 512x16B) --
#pragma unroll
    for (int i = 0; i < 2; ++i) {
      const int idx = tid + i * 256;           // 0..511
      const int row = idx >> 2;
      const int ch  = idx & 3;                 // 16B chunk in row
      const int c8  = ch << 3;                 // half offset in row
      async_b128((unsigned)(size_t)&As[row][ch << 2],
                 A16 + (size_t)(m0 + row) * K + k0 + c8);
      async_b128((unsigned)(size_t)&Bs[row][ch << 2],
                 Bt16 + (size_t)(n0 + row) * K + k0 + c8);
    }
    wait_async0();
    __syncthreads();                 // copies from all waves visible

    // ---- fragments (8 x ds_read_b32 each) + 8 WMMAs per wave ----
    FragH af[2];
#pragma unroll
    for (int mi = 0; mi < 2; ++mi) {
      const unsigned* arow = As[wm * 32 + mi * 16 + l16];
#pragma unroll
      for (int v = 0; v < 8; ++v) {
        // A layout: VGPR v <- K=(v&3)*2+16*(v>>2)+8*half => dword (v&3)+8*(v>>2)+4*half
        af[mi].u[v] = arow[(v & 3) + ((v >> 2) << 3) + (half << 2)];
      }
    }
    FragH bf[4];
#pragma unroll
    for (int ni = 0; ni < 4; ++ni) {
      const unsigned* brow = Bs[wn * 64 + ni * 16 + l16];
#pragma unroll
      for (int v = 0; v < 8; ++v) {
        // B layout: VGPR v <- K=(v&3)*2+16*(v>>2)+8*half => dword (v&3)+8*(v>>2)+4*half
        bf[ni].u[v] = brow[(v & 3) + ((v >> 2) << 3) + (half << 2)];
      }
    }
#pragma unroll
    for (int mi = 0; mi < 2; ++mi)
#pragma unroll
      for (int ni = 0; ni < 4; ++ni)
        acc[mi][ni] = wmma_f16(af[mi].h, bf[ni].h, acc[mi][ni]);
  }

  // ---- epilogue: bias / relu / store (C layout: VGPR r -> M=r(+8), N=lane) --
#pragma unroll
  for (int mi = 0; mi < 2; ++mi) {
#pragma unroll
    for (int ni = 0; ni < 4; ++ni) {
      const int col = n0 + wn * 64 + ni * 16 + l16;
      const float bsv = bias ? bias[col] : 0.0f;
      float* cp = C + (size_t)(m0 + wm * 32 + mi * 16 + (half << 3)) * N + col;
#pragma unroll
      for (int r = 0; r < 8; ++r) {
        float vvv = acc[mi][ni][r] + bsv;
        if (relu) vvv = fmaxf(vvv, 0.0f);
        cp[(size_t)r * N] = vvv;
      }
    }
  }
}

// --------------------------- fused attention -------------------------------
// One wave per (b, h, 16-query tile).  QK^T and P@V through WMMA, softmax over
// the 512-wide score row staged in LDS.  Writes the reference's "scrambled
// transpose" layout: Y[b][h*64+dh][q].
__global__ __launch_bounds__(32)
void attn_kernel(const float* __restrict__ Q, const float* __restrict__ Kb,
                 const float* __restrict__ Vb, const int* __restrict__ pad,
                 float* __restrict__ Yout, int causal) {
  __shared__ float S[16][520];
  const int b = blockIdx.x / Hh;
  const int h = blockIdx.x % Hh;
  const int qbase = blockIdx.y * 16;
  const int lane = threadIdx.x;
  const int half = lane >> 4;
  const int l16  = lane & 15;

  // Q fragments (16 x 64 split into two 16x32 chunks)
  FragH qf[2];
  {
    const int row = qbase + l16;
    const float* qrow = Q + ((size_t)b * Tt + row) * Dm + h * DHd;
#pragma unroll
    for (int kc = 0; kc < 2; ++kc)
#pragma unroll
      for (int v = 0; v < 8; ++v) {
        int kk = kc * 32 + ((v & 3) << 1) + ((v >> 2) << 4) + (half << 3);
        qf[kc].u[v] = pk2(qrow[kk], qrow[kk + 1]);
      }
  }

  // ---- S = Q K^T / sqrt(64), masked, into LDS ----
  for (int kt = 0; kt < Tt / 16; ++kt) {
    v8f s = {};
    const int key = kt * 16 + l16;
    const float* krow = Kb + ((size_t)b * Tt + key) * Dm + h * DHd;
#pragma unroll
    for (int kc = 0; kc < 2; ++kc) {
      FragH kf;
#pragma unroll
      for (int v = 0; v < 8; ++v) {
        int kk = kc * 32 + ((v & 3) << 1) + ((v >> 2) << 4) + (half << 3);
        kf.u[v] = pk2(krow[kk], krow[kk + 1]);
      }
      s = wmma_f16(qf[kc].h, kf.h, s);
    }
    const int keep = pad[b * Tt + key];
#pragma unroll
    for (int r = 0; r < 8; ++r) {
      const int m  = r + (half << 3);
      const int qq = qbase + m;
      float val = s[r] * 0.125f;
      val = (causal && key > qq) ? NEG_BIG : val;
      val = keep ? val : NEG_BIG;
      S[m][key] = val;
    }
  }
  __syncthreads();

  // ---- softmax per row (wave-cooperative) ----
  for (int r = 0; r < 16; ++r) {
    float mx = NEG_BIG;
    for (int c = lane; c < Tt; c += 32) mx = fmaxf(mx, S[r][c]);
#pragma unroll
    for (int o = 16; o > 0; o >>= 1) mx = fmaxf(mx, __shfl_xor(mx, o, 32));
    float sum = 0.0f;
    for (int c = lane; c < Tt; c += 32) {
      float e = __expf(S[r][c] - mx);
      S[r][c] = e;
      sum += e;
    }
#pragma unroll
    for (int o = 16; o > 0; o >>= 1) sum += __shfl_xor(sum, o, 32);
    const float inv = (sum > 0.0f) ? (1.0f / sum) : 0.0f;
    for (int c = lane; c < Tt; c += 32) S[r][c] *= inv;
  }
  __syncthreads();

  // ---- Y = P @ V ----
  v8f o4[4] = {};
  for (int kt = 0; kt < Tt / 32; ++kt) {
    FragH pa;
#pragma unroll
    for (int v = 0; v < 8; ++v) {
      int kk = kt * 32 + ((v & 3) << 1) + ((v >> 2) << 4) + (half << 3);
      pa.u[v] = pk2(S[l16][kk], S[l16][kk + 1]);
    }
#pragma unroll
    for (int nf = 0; nf < 4; ++nf) {
      FragH vf;
      const int col = nf * 16 + l16;
#pragma unroll
      for (int v = 0; v < 8; ++v) {
        const int key = kt * 32 + ((v & 3) << 1) + ((v >> 2) << 4) + (half << 3);
        const float* vr = Vb + ((size_t)b * Tt + key) * Dm + h * DHd + col;
        vf.u[v] = pk2(vr[0], vr[Dm]);
      }
      o4[nf] = wmma_f16(pa.h, vf.h, o4[nf]);
    }
  }
#pragma unroll
  for (int nf = 0; nf < 4; ++nf) {
    const int dh = nf * 16 + l16;
    float* yp = Yout + ((size_t)b * Tt + h * DHd + dh) * Dm + qbase + (half << 3);
#pragma unroll
    for (int r = 0; r < 8; ++r) yp[r] = o4[nf][r];
  }
}

// ------------------------------- host side ---------------------------------
struct Scratch {
  _Float16* a16;
  _Float16* b16;
};

// bT==0: Bsrc is fp32 [K,N] (transpose-convert); bT==1: Bsrc is fp32 [N,K].
static void gemm(hipStream_t s, Scratch sc, const float* A, const float* Bsrc,
                 int bT, const float* bias, int relu, float* C,
                 int M, int N, int K) {
  cvt_kernel<<<(size_t)M * K / 1024, 256, 0, s>>>(A, sc.a16);
  if (bT)
    cvt_kernel<<<(size_t)N * K / 1024, 256, 0, s>>>(Bsrc, sc.b16);
  else
    cvt_tr_kernel<<<dim3(N / 32, K / 32), 256, 0, s>>>(Bsrc, sc.b16, K, N);
  dim3 grid(N / BN, M / BM);
  gemm_kernel<<<grid, 256, 0, s>>>(sc.a16, sc.b16, bias, relu, C, M, N, K);
}

extern "C" void kernel_launch(void* const* d_in, const int* in_sizes, int n_in,
                              void* d_out, int out_size, void* d_ws, size_t ws_size,
                              hipStream_t stream) {
  const int*   src_ids   = (const int*)d_in[0];
  const int*   tgt_ids   = (const int*)d_in[1];
  const int*   src_pad   = (const int*)d_in[2];
  const int*   tgt_pad   = (const int*)d_in[3];
  const float* emb       = (const float*)d_in[4];
  const float* enc_w     = (const float*)d_in[5];   // [NL,4,D,D]
  const float* enc_ln    = (const float*)d_in[6];   // [NL,2,2,D]
  const float* enc_w1    = (const float*)d_in[7];   // [NL,D,4D]
  const float* enc_b1    = (const float*)d_in[8];
  const float* enc_w2    = (const float*)d_in[9];   // [NL,4D,D]
  const float* enc_b2    = (const float*)d_in[10];
  const float* dsw       = (const float*)d_in[11];  // [NL,4,D,D]
  const float* dcw       = (const float*)d_in[12];  // [NL,4,D,D]
  const float* dec_ln    = (const float*)d_in[13];  // [NL,3,2,D]
  const float* dec_w1    = (const float*)d_in[14];
  const float* dec_b1    = (const float*)d_in[15];
  const float* dec_w2    = (const float*)d_in[16];
  const float* dec_b2    = (const float*)d_in[17];

  const int M = Bb * Tt;                       // 4096 rows
  const size_t ACT = (size_t)M * Dm;           // 2M floats (8 MB)
  float* ws    = (float*)d_ws;                 // needs ~142 MB scratch
  float* enc_h = ws + 0 * ACT;
  float* dec_h = ws + 1 * ACT;
  float* tbuf  = ws + 2 * ACT;
  float* ybuf  = ws + 3 * ACT;
  float* qb    = ws + 4 * ACT;
  float* kb    = ws + 5 * ACT;
  float* vb    = ws + 6 * ACT;
  float* ffn   = ws + 7 * ACT;                 // 4*ACT floats
  Scratch sc;
  sc.a16 = (_Float16*)(ws + 11 * ACT);         // up to 4096x2048 halves
  sc.b16 = sc.a16 + (size_t)4096 * 2048;       // up to 32000x512 halves

  const size_t WDD = (size_t)Dm * Dm;          // 262144
  const dim3 agrid(Bb * Hh, Tt / 16);

  // ---- embeddings + positional encoding ----
  embed_kernel<<<M, 256, 0, stream>>>(src_ids, emb, enc_h);
  embed_kernel<<<M, 256, 0, stream>>>(tgt_ids, emb, dec_h);

  // ---- encoder ----
  for (int l = 0; l < NLl; ++l) {
    const float* w  = enc_w + (size_t)l * 4 * WDD;
    const float* g0 = enc_ln + (size_t)((l * 2 + 0) * 2) * Dm;
    const float* g1 = enc_ln + (size_t)((l * 2 + 1) * 2) * Dm;
    gemm(stream, sc, enc_h, w + 0 * WDD, 0, nullptr, 0, qb, M, Dm, Dm);
    gemm(stream, sc, enc_h, w + 1 * WDD, 0, nullptr, 0, kb, M, Dm, Dm);
    gemm(stream, sc, enc_h, w + 2 * WDD, 0, nullptr, 0, vb, M, Dm, Dm);
    attn_kernel<<<agrid, 32, 0, stream>>>(qb, kb, vb, src_pad, ybuf, 0);
    gemm(stream, sc, ybuf, w + 3 * WDD, 0, nullptr, 0, tbuf, M, Dm, Dm);
    ln_add_kernel<<<M, 256, 0, stream>>>(enc_h, tbuf, g0, g0 + Dm, tbuf);
    gemm(stream, sc, tbuf, enc_w1 + (size_t)l * Dm * 4 * Dm, 0,
         enc_b1 + (size_t)l * 4 * Dm, 1, ffn, M, 4 * Dm, Dm);
    gemm(stream, sc, ffn, enc_w2 + (size_t)l * 4 * Dm * Dm, 0,
         enc_b2 + (size_t)l * Dm, 0, ybuf, M, Dm, 4 * Dm);
    ln_add_kernel<<<M, 256, 0, stream>>>(tbuf, ybuf, g1, g1 + Dm, enc_h);
  }

  // ---- decoder ----
  for (int l = 0; l < NLl; ++l) {
    const float* wsf = dsw + (size_t)l * 4 * WDD;
    const float* wcr = dcw + (size_t)l * 4 * WDD;
    const float* g0 = dec_ln + (size_t)((l * 3 + 0) * 2) * Dm;
    const float* g1 = dec_ln + (size_t)((l * 3 + 1) * 2) * Dm;
    const float* g2 = dec_ln + (size_t)((l * 3 + 2) * 2) * Dm;
    // masked self-attention
    gemm(stream, sc, dec_h, wsf + 0 * WDD, 0, nullptr, 0, qb, M, Dm, Dm);
    gemm(stream, sc, dec_h, wsf + 1 * WDD, 0, nullptr, 0, kb, M, Dm, Dm);
    gemm(stream, sc, dec_h, wsf + 2 * WDD, 0, nullptr, 0, vb, M, Dm, Dm);
    attn_kernel<<<agrid, 32, 0, stream>>>(qb, kb, vb, tgt_pad, ybuf, 1);
    gemm(stream, sc, ybuf, wsf + 3 * WDD, 0, nullptr, 0, tbuf, M, Dm, Dm);
    ln_add_kernel<<<M, 256, 0, stream>>>(dec_h, tbuf, g0, g0 + Dm, tbuf);
    // cross-attention (K/V from encoder output)
    gemm(stream, sc, tbuf, wcr + 0 * WDD, 0, nullptr, 0, qb, M, Dm, Dm);
    gemm(stream, sc, enc_h, wcr + 1 * WDD, 0, nullptr, 0, kb, M, Dm, Dm);
    gemm(stream, sc, enc_h, wcr + 2 * WDD, 0, nullptr, 0, vb, M, Dm, Dm);
    attn_kernel<<<agrid, 32, 0, stream>>>(qb, kb, vb, src_pad, ybuf, 0);
    gemm(stream, sc, ybuf, wcr + 3 * WDD, 0, nullptr, 0, dec_h, M, Dm, Dm);
    ln_add_kernel<<<M, 256, 0, stream>>>(tbuf, dec_h, g1, g1 + Dm, tbuf);
    // FFN
    gemm(stream, sc, tbuf, dec_w1 + (size_t)l * Dm * 4 * Dm, 0,
         dec_b1 + (size_t)l * 4 * Dm, 1, ffn, M, 4 * Dm, Dm);
    gemm(stream, sc, ffn, dec_w2 + (size_t)l * 4 * Dm * Dm, 0,
         dec_b2 + (size_t)l * Dm, 0, dec_h, M, Dm, 4 * Dm);
    ln_add_kernel<<<M, 256, 0, stream>>>(tbuf, dec_h, g2, g2 + Dm, dec_h);
  }

  // ---- tied output projection: dec @ emb.T  -> [4096, 32000] ----
  gemm(stream, sc, dec_h, emb, /*bT=*/1, nullptr, 0, (float*)d_out, M, Vv, Dm);
}
